// HashSideOut_5025111736932
// MI455X (gfx1250) — matrix operand order repaired and checked by reading, not compiled, not run on hardware
//
#include <hip/hip_runtime.h>
#include <hip/hip_bf16.h>

typedef __attribute__((ext_vector_type(16))) _Float16 v16h;
typedef __attribute__((ext_vector_type(8)))  float    v8f;

#define TABLE_NUM  16
#define TABLE_SIZE 65536
#define IN_CH      32
#define OUT_CH     3
#define RESOLUTION 512
#define TILE       16
#define HALO       18
#define NPIX_HALO  (HALO * HALO)          /* 324 */
#define NTASK      (NPIX_HALO * TABLE_NUM) /* 5184 */
#define STYLE_DIM  512
#define PRIME_Y    2654435761u

// RES_MIN * growth^l with growth = 2^(1/3)  (RES 16 -> 512 over 16 levels)
__device__ __constant__ float g_res_levels[TABLE_NUM] = {
    16.0f,       20.1587368f, 25.3984168f, 32.0f,
    40.3174736f, 50.7968337f, 64.0f,       80.6349472f,
    101.593667f, 128.0f,      161.269894f, 203.187335f,
    256.0f,      322.539789f, 406.374669f, 512.0f};

union V16HU { _Float16 e[16]; v16h v; };
union AFragU { uint4 u[2]; v16h v; };

// ---------------------------------------------------------------------------
// Kernel 1: styles[b][c] = (s[b] . affine_w[c]) / sqrt(512) + affine_b[c]
//           dcoef[b][o]  = rsqrt(sum_{i,hw} (w[o][i][hw] * styles[b][i])^2 + 1e-8)
// ws layout: [0..127] styles, [128..139] dcoef
// ---------------------------------------------------------------------------
__global__ void hash_styles_kernel(const float* __restrict__ s,
                                   const float* __restrict__ weight,
                                   const float* __restrict__ affine_w,
                                   const float* __restrict__ affine_b,
                                   float* __restrict__ ws) {
    const int t = threadIdx.x;
    __shared__ float st[128];
    if (t < 128) {
        const int b = t >> 5, c = t & 31;
        const float* sv = s + (size_t)b * STYLE_DIM;
        const float* av = affine_w + (size_t)c * STYLE_DIM;
        float acc = 0.f;
        for (int k = 0; k < STYLE_DIM; ++k) acc = fmaf(sv[k], av[k], acc);
        const float val = acc * 0.044194173824159216f + affine_b[c];  // 1/sqrt(512)
        st[t] = val;
        ws[t] = val;
    }
    __syncthreads();
    if (t < 4 * OUT_CH) {
        const int b = t / OUT_CH, o = t % OUT_CH;
        float sum = 0.f;
        for (int i = 0; i < IN_CH; ++i) {
            const float si = st[b * IN_CH + i];
            for (int k = 0; k < 9; ++k) {
                const float w = weight[(o * IN_CH + i) * 9 + k] * si;
                sum = fmaf(w, w, sum);
            }
        }
        ws[128 + t] = rsqrtf(sum + 1e-8f);
    }
}

// ---------------------------------------------------------------------------
// Kernel 2: fused hash-grid lookup (-> LDS, f16, style-modulated) + 3x3 conv
// via v_wmma_f32_16x16x32_f16, demod + bias + clamp, store fp32.
// Grid: (32, 32, B), block: 256 threads = 8 waves.
// ---------------------------------------------------------------------------
__global__ __launch_bounds__(256)
void hash_conv_wmma_kernel(const float* __restrict__ x,
                           const float* __restrict__ weight,
                           const float* __restrict__ bias,
                           const float* __restrict__ ws,
                           float* __restrict__ out) {
    const int tx = blockIdx.x, ty = blockIdx.y, b = blockIdx.z;
    const int tid = threadIdx.x;
    const int lane = tid & 31, wave = tid >> 5;

    // halo feature tile, f16, [pix 0..323][ch 0..31]
    __shared__ __align__(16) _Float16 sh[NPIX_HALO * IN_CH];  // 20736 B

    // ---------------- Phase 1: hash-grid features into LDS -----------------
    const int gx0 = tx * TILE - 1;
    const int gy0 = ty * TILE - 1;
    const float inv_res = 1.0f / (float)RESOLUTION;

    for (int task = tid; task < NTASK; task += 256) {
        const int pix = task >> 4;
        const int lvl = task & 15;
        const int hy = pix / HALO;
        const int hx = pix - hy * HALO;
        const int gy = gy0 + hy, gx = gx0 + hx;

        unsigned pack = 0u;  // zero features outside the image (SAME padding)
        if ((unsigned)gy < RESOLUTION && (unsigned)gx < RESOLUTION) {
            const float res  = g_res_levels[lvl];
            const float posx = ((float)gx + 0.5f) * inv_res * res;
            const float posy = ((float)gy + 0.5f) * inv_res * res;
            const float p0x = floorf(posx), p0y = floorf(posy);
            const float fx = posx - p0x, fy = posy - p0y;
            const unsigned ux = (unsigned)p0x, uy = (unsigned)p0y;

            const float* tab = x + (size_t)(b * TABLE_NUM + lvl) * (TABLE_SIZE * 2);
            const unsigned hy0 = uy * PRIME_Y;
            const unsigned hy1 = (uy + 1u) * PRIME_Y;
            const unsigned i00 = ( ux        ^ hy0) & (TABLE_SIZE - 1);
            const unsigned i10 = ((ux + 1u)  ^ hy0) & (TABLE_SIZE - 1);
            const unsigned i01 = ( ux        ^ hy1) & (TABLE_SIZE - 1);
            const unsigned i11 = ((ux + 1u)  ^ hy1) & (TABLE_SIZE - 1);

            const float2 f00 = *(const float2*)(tab + 2u * i00);
            const float2 f10 = *(const float2*)(tab + 2u * i10);
            const float2 f01 = *(const float2*)(tab + 2u * i01);
            const float2 f11 = *(const float2*)(tab + 2u * i11);

            const float w00 = (1.f - fx) * (1.f - fy);
            const float w10 = fx * (1.f - fy);
            const float w01 = (1.f - fx) * fy;
            const float w11 = fx * fy;

            const float f0 = w00 * f00.x + w10 * f10.x + w01 * f01.x + w11 * f11.x;
            const float f1 = w00 * f00.y + w10 * f10.y + w01 * f01.y + w11 * f11.y;

            // style modulation (xs = h * styles) folded into the LDS tile
            const float s0 = ws[b * IN_CH + 2 * lvl];
            const float s1 = ws[b * IN_CH + 2 * lvl + 1];
            union { _Float16 h[2]; unsigned u; } pk;
            pk.h[0] = (_Float16)(f0 * s0);
            pk.h[1] = (_Float16)(f1 * s1);
            pack = pk.u;
        }
        *(unsigned*)&sh[pix * IN_CH + 2 * lvl] = pack;
    }
    __syncthreads();

    // ---------------- Phase 2: 3x3 conv as 9x WMMA accumulation ------------
    // GEMM per output row: M=16 pixels, N=16 (3 real out-ch), K=9 taps x 32 ch
    const int n    = lane & 15;                 // output-channel column
    const int kofs = (lane >= 16) ? 16 : 0;     // B-fragment K base for this lane is kofs/2

    // B fragments: one 32x16 (KxN) tile per tap, weights as f16, cols >=3 zero
    v16h bf[9];
#pragma unroll
    for (int t = 0; t < 9; ++t) {
        V16HU bv;
#pragma unroll
        for (int i = 0; i < 16; ++i) {
            float wv = 0.f;
            if (n < OUT_CH) wv = weight[(n * IN_CH + ((kofs >> 1) + i + ((i & 8) ? 8 : 0))) * 9 + t];
            bv.e[i] = (_Float16)wv;
        }
        bf[t] = bv.v;
    }

    float dco = 0.f, bi = 0.f;
    if (n < OUT_CH) {
        dco = ws[128 + b * OUT_CH + n];
        bi  = bias[n];
    }

    const int abase = (lane >= 16) ? 8 : 0;  // A-fragment channel base per ISA layout
    const int ox_a  = lane & 15;             // A row (pixel column) this lane supplies

#pragma unroll
    for (int g = 0; g < 2; ++g) {
        const int oy = wave * 2 + g;  // output row inside the tile (8 waves x 2)
        v8f c = {};
#pragma unroll
        for (int t = 0; t < 9; ++t) {
            const int dy = t / 3, dx = t % 3;
            const int hpix = (oy + dy) * HALO + (ox_a + dx);
            // two contiguous 8-half chunks: ch [abase, abase+8) and [abase+16, abase+24)
            const uint4* p = (const uint4*)&sh[hpix * IN_CH + abase];
            AFragU av;
            av.u[0] = p[0];
            av.u[1] = p[2];
            c = __builtin_amdgcn_wmma_f32_16x16x32_f16(
                    /*neg_a=*/false, av.v, /*neg_b=*/false, bf[t],
                    /*c_mod=*/(short)0, c, /*reuse_a=*/false, /*reuse_b=*/false);
        }
        // D layout: VGPR r, lanes<16 -> pixel r, lanes>=16 -> pixel r+8; N = lane&15
        if (n < OUT_CH) {
            const int gy = ty * TILE + oy;
            const int oxb = tx * TILE + ((lane >= 16) ? 8 : 0);
            const size_t rowbase = (((size_t)b * OUT_CH + n) * RESOLUTION + gy) * RESOLUTION;
#pragma unroll
            for (int r = 0; r < 8; ++r) {
                float v = fmaf(c[r], dco, bi);
                v = fminf(fmaxf(v, -256.0f), 256.0f);
                out[rowbase + oxb + r] = v;
            }
        }
    }
}

// ---------------------------------------------------------------------------
extern "C" void kernel_launch(void* const* d_in, const int* in_sizes, int n_in,
                              void* d_out, int out_size, void* d_ws, size_t ws_size,
                              hipStream_t stream) {
    const float* x        = (const float*)d_in[0];  // (4,16,131072)
    const float* s        = (const float*)d_in[1];  // (4,512)
    const float* weight   = (const float*)d_in[2];  // (3,32,3,3)
    const float* affine_w = (const float*)d_in[3];  // (32,512)
    const float* affine_b = (const float*)d_in[4];  // (32,)
    const float* bias     = (const float*)d_in[5];  // (3,)
    float* out = (float*)d_out;                     // (4,3,512,512)
    float* ws  = (float*)d_ws;                      // styles[128] + dcoef[12]

    hash_styles_kernel<<<1, 128, 0, stream>>>(s, weight, affine_w, affine_b, ws);

    dim3 grid(RESOLUTION / TILE, RESOLUTION / TILE, 4);
    hash_conv_wmma_kernel<<<grid, 256, 0, stream>>>(x, weight, bias, ws, out);
}
